// HemaGraph_88055419503158
// MI455X (gfx1250) — matrix operand order repaired and checked by reading, not compiled, not run on hardware
//
#include <hip/hip_runtime.h>
#include <hip/hip_bf16.h>
#include <math.h>

#define HEADS 8
#define NEG_SLOPE 0.2f

typedef __attribute__((ext_vector_type(16))) _Float16 v16h;
typedef __attribute__((ext_vector_type(8)))  _Float16 v8h;
typedef __attribute__((ext_vector_type(8)))  float    v8f;

// ---------------------------------------------------------------------------
// Pack fp32 activations -> zero-padded row-major f16 [M, Kpad], Kpad % 32 == 0
// ---------------------------------------------------------------------------
__global__ void pack_a_f16(const float* __restrict__ X, _Float16* __restrict__ Apad,
                           int M, int K, int Kpad) {
  const size_t i = (size_t)blockIdx.x * blockDim.x + threadIdx.x;
  if (i >= (size_t)M * Kpad) return;
  const int k = (int)(i % Kpad);
  const int m = (int)(i / Kpad);
  Apad[i] = (k < K) ? (_Float16)X[(size_t)m * K + k] : (_Float16)0.f;
}

// ---------------------------------------------------------------------------
// Pack fp32 W[K,Nc] -> B-fragment layout [tile16][kstep][lane][16 halves]
// element j of lane L: K = kstep*32 + (L<16?0:16) + j, col = tile*16 + (L&15)
// Zero-padded in K and N (Ntile16p is a multiple of 4 -> 64-col strips).
// ---------------------------------------------------------------------------
__global__ void pack_b_f16(const float* __restrict__ W, _Float16* __restrict__ Bpack,
                           int K, int Nc, int Ksteps, int Ntile16p) {
  const size_t i = (size_t)blockIdx.x * blockDim.x + threadIdx.x;
  if (i >= (size_t)Ntile16p * Ksteps * 512) return;
  const int j    = (int)(i & 15);
  const int lane = (int)((i >> 4) & 31);
  const size_t g = i >> 9;                       // tile*Ksteps + kstep
  const int ks   = (int)(g % Ksteps);
  const int tile = (int)(g / Ksteps);
  const int k    = ks * 32 + ((lane < 16) ? 0 : 16) + j;
  const int col  = tile * 16 + (lane & 15);
  Bpack[i] = (k < K && col < Nc) ? (_Float16)W[(size_t)k * Nc + col] : (_Float16)0.f;
}

// ---------------------------------------------------------------------------
// WMMA GEMM on packed f16: Hout[M,Nc] = Apad * W. One wave: 32(M) x 64(N).
// Inner loop: 4 b128 A loads + 8 b128 B loads + 8 WMMAs per K-step. No branches.
// ---------------------------------------------------------------------------
__global__ __launch_bounds__(128)
void gat_gemm_wmma(const _Float16* __restrict__ Apad,
                   const _Float16* __restrict__ Bpack,
                   float* __restrict__ Hout,
                   int M, int Kpad, int Nc) {
  const int lane  = threadIdx.x & 31;
  const int wave  = threadIdx.x >> 5;
  const int mbase = (blockIdx.x * 4 + wave) * 32;
  if (mbase >= M) return;                      // uniform per wave
  const int ntile0 = blockIdx.y * 4;           // 64-column strip
  const int Ksteps = Kpad >> 5;

  v8f acc[2][4];
#pragma unroll
  for (int i = 0; i < 2; ++i)
#pragma unroll
    for (int t = 0; t < 4; ++t) acc[i][t] = v8f{};

  const int koffA = (lane < 16) ? 0 : 16;      // A: upper half-wave holds K+16
  int rowA0 = mbase + (lane & 15);
  int rowA1 = rowA0 + 16;
  rowA0 = (rowA0 < M) ? rowA0 : (M - 1);       // clamp; garbage rows never stored
  rowA1 = (rowA1 < M) ? rowA1 : (M - 1);

  const _Float16* aptr0 = Apad + (size_t)rowA0 * Kpad + koffA;
  const _Float16* aptr1 = Apad + (size_t)rowA1 * Kpad + koffA;
  const _Float16* bbase = Bpack + (size_t)lane * 16;

  for (int ks = 0; ks < Ksteps; ++ks) {
    const int k0 = ks << 5;
    union { v16h v; v8h h[2]; } a0, a1;
    a0.h[0] = *(const v8h*)(aptr0 + k0);
    a0.h[1] = *(const v8h*)(aptr0 + k0 + 8);
    a1.h[0] = *(const v8h*)(aptr1 + k0);
    a1.h[1] = *(const v8h*)(aptr1 + k0 + 8);
#pragma unroll
    for (int t = 0; t < 4; ++t) {
      union { v16h v; v8h h[2]; } b;
      const _Float16* bp = bbase + ((size_t)(ntile0 + t) * Ksteps + ks) * 512;
      b.h[0] = *(const v8h*)bp;
      b.h[1] = *(const v8h*)(bp + 8);
      acc[0][t] = __builtin_amdgcn_wmma_f32_16x16x32_f16(
          false, a0.v, false, b.v, (short)0, acc[0][t], false, false);
      acc[1][t] = __builtin_amdgcn_wmma_f32_16x16x32_f16(
          false, a1.v, false, b.v, (short)0, acc[1][t], false, false);
    }
  }

  const int colL = lane & 15;
  const int rsel = (lane < 16) ? 0 : 8;
#pragma unroll
  for (int i = 0; i < 2; ++i) {
    const int rbase = mbase + i * 16 + rsel;
#pragma unroll
    for (int t = 0; t < 4; ++t) {
      const int ncol = (ntile0 + t) * 16 + colL;
      if (ncol >= Nc) continue;
#pragma unroll
      for (int r = 0; r < 8; ++r) {
        const int row = rbase + r;
        if (row < M) Hout[(size_t)row * Nc + ncol] = acc[i][t][r];
      }
    }
  }
}

// ---------------------------------------------------------------------------
// Attention logits: als[n,h] = <h[n,h,:], a_src[h,:]>, ald likewise.
// ---------------------------------------------------------------------------
__global__ void attn_logits(const float* __restrict__ H,
                            const float* __restrict__ a_src,
                            const float* __restrict__ a_dst,
                            float* __restrict__ als, float* __restrict__ ald,
                            int N, int C) {
  const int idx = blockIdx.x * blockDim.x + threadIdx.x;   // n*HEADS + h
  if (idx >= N * HEADS) return;
  const int n = idx / HEADS, h = idx % HEADS;
  const float* hp = H + (size_t)n * HEADS * C + (size_t)h * C;
  const float* sp = a_src + h * C;
  const float* dp = a_dst + h * C;
  float s = 0.f, d = 0.f;
  for (int c = 0; c < C; ++c) { const float v = hp[c]; s += v * sp[c]; d += v * dp[c]; }
  als[idx] = s; ald[idx] = d;
}

__global__ void fill_f32(float* __restrict__ p, float v, size_t n) {
  const size_t i = (size_t)blockIdx.x * blockDim.x + threadIdx.x;
  if (i < n) p[i] = v;
}

__device__ __forceinline__ void atomicMaxF32(float* addr, float val) {
  if (val >= 0.f) atomicMax((int*)addr, __float_as_int(val));
  else            atomicMin((unsigned int*)addr, (unsigned int)__float_as_int(val));
}

__device__ __forceinline__ void edge_nodes(int e, int E_in,
                                           const int* __restrict__ esrc,
                                           const int* __restrict__ edst,
                                           int& s, int& d) {
  if (e < E_in) { s = esrc[e]; d = edst[e]; }
  else          { s = d = e - E_in; }          // self-loops appended
}

__global__ void edge_max(const int* __restrict__ esrc, const int* __restrict__ edst,
                         int E_in, int Etot,
                         const float* __restrict__ als, const float* __restrict__ ald,
                         float* __restrict__ emax) {
  const int idx = blockIdx.x * blockDim.x + threadIdx.x;   // e*HEADS + h
  if (idx >= Etot * HEADS) return;
  const int e = idx / HEADS, h = idx % HEADS;
  int s, d; edge_nodes(e, E_in, esrc, edst, s, d);
  float x = als[s * HEADS + h] + ald[d * HEADS + h];
  x = (x >= 0.f) ? x : NEG_SLOPE * x;                      // leaky_relu
  atomicMaxF32(&emax[d * HEADS + h], x);
}

__global__ void edge_expsum(const int* __restrict__ esrc, const int* __restrict__ edst,
                            int E_in, int Etot,
                            const float* __restrict__ als, const float* __restrict__ ald,
                            const float* __restrict__ emax,
                            float* __restrict__ denom, float* __restrict__ exb) {
  const int idx = blockIdx.x * blockDim.x + threadIdx.x;
  if (idx >= Etot * HEADS) return;
  const int e = idx / HEADS, h = idx % HEADS;
  int s, d; edge_nodes(e, E_in, esrc, edst, s, d);
  float x = als[s * HEADS + h] + ald[d * HEADS + h];
  x = (x >= 0.f) ? x : NEG_SLOPE * x;
  const float ex = __expf(x - emax[d * HEADS + h]);
  exb[idx] = ex;
  atomicAdd(&denom[d * HEADS + h], ex);
}

// One block per edge; threads stride the H*C channels; atomic scatter to dst.
__global__ void aggregate(const int* __restrict__ esrc, const int* __restrict__ edst,
                          int E_in,
                          const float* __restrict__ H, const float* __restrict__ exb,
                          const float* __restrict__ denom,
                          float* __restrict__ out, int C) {
  const int e = blockIdx.x;
  int s, d; edge_nodes(e, E_in, esrc, edst, s, d);
  const int HC = HEADS * C;
  const float* hs = H + (size_t)s * HC;
  float* od = out + (size_t)d * HC;
  for (int c = threadIdx.x; c < HC; c += blockDim.x) {
    const int h = c / C;
    const float alpha = exb[(size_t)e * HEADS + h] / (denom[d * HEADS + h] + 1e-16f);
    atomicAdd(&od[c], hs[c] * alpha);
  }
}

__global__ void bias_relu(float* __restrict__ X, const float* __restrict__ b,
                          size_t total, int HC) {
  const size_t i = (size_t)blockIdx.x * blockDim.x + threadIdx.x;
  if (i < total) X[i] = fmaxf(X[i] + b[i % HC], 0.f);
}

// Layer 3: mean over heads (+b3), then log_softmax over 5 classes -> d_out.
__global__ void head_mean_logsoftmax(const float* __restrict__ agg,
                                     const float* __restrict__ b3,
                                     float* __restrict__ out, int N) {
  const int n = blockIdx.x * blockDim.x + threadIdx.x;
  if (n >= N) return;
  const int C = 5;
  float v[5];
#pragma unroll
  for (int c = 0; c < C; ++c) {
    float s = 0.f;
#pragma unroll
    for (int h = 0; h < HEADS; ++h) s += agg[(size_t)n * HEADS * C + h * C + c];
    v[c] = s * (1.0f / HEADS) + b3[c];
  }
  float m = v[0];
#pragma unroll
  for (int c = 1; c < C; ++c) m = fmaxf(m, v[c]);
  float se = 0.f;
#pragma unroll
  for (int c = 0; c < C; ++c) se += __expf(v[c] - m);
  const float lse = m + __logf(se);
#pragma unroll
  for (int c = 0; c < C; ++c) out[(size_t)n * C + c] = v[c] - lse;
}

// ---------------------------------------------------------------------------
// Host-side per-layer launcher
// ---------------------------------------------------------------------------
static void gat_layer_launch(const float* Xin, int Fin, int C,
                             const float* W, const float* a_s, const float* a_d,
                             _Float16* Apad, _Float16* Bpack,
                             float* hbuf, float* obuf,
                             float* als, float* ald, float* emax, float* denom,
                             float* exb,
                             const int* esrc, const int* edst, int E_in, int N,
                             hipStream_t stream) {
  const int HC   = HEADS * C;
  const int Etot = E_in + N;
  const int Kpad = ((Fin + 31) / 32) * 32;
  const int Ksteps = Kpad / 32;
  const int Ntile16p = ((HC + 63) / 64) * 4;   // multiple of 4 tiles (64 cols)

  // Pack operands to f16
  const size_t na = (size_t)N * Kpad;
  pack_a_f16<<<(unsigned)((na + 255) / 256), 256, 0, stream>>>(Xin, Apad, N, Fin, Kpad);
  const size_t nb = (size_t)Ntile16p * Ksteps * 512;
  pack_b_f16<<<(unsigned)((nb + 255) / 256), 256, 0, stream>>>(W, Bpack, Fin, HC,
                                                               Ksteps, Ntile16p);

  // WMMA GEMM
  const int mstrips = (N + 31) / 32;
  dim3 ggrid((mstrips + 3) / 4, Ntile16p / 4);
  gat_gemm_wmma<<<ggrid, 128, 0, stream>>>(Apad, Bpack, hbuf, N, Kpad, HC);

  attn_logits<<<(N * HEADS + 255) / 256, 256, 0, stream>>>(hbuf, a_s, a_d, als, ald, N, C);

  const size_t nh = (size_t)N * HEADS;
  fill_f32<<<(unsigned)((nh + 255) / 256), 256, 0, stream>>>(emax, -INFINITY, nh);
  fill_f32<<<(unsigned)((nh + 255) / 256), 256, 0, stream>>>(denom, 0.f, nh);
  const size_t nhc = (size_t)N * HC;
  fill_f32<<<(unsigned)((nhc + 255) / 256), 256, 0, stream>>>(obuf, 0.f, nhc);

  edge_max<<<(Etot * HEADS + 255) / 256, 256, 0, stream>>>(esrc, edst, E_in, Etot, als, ald, emax);
  edge_expsum<<<(Etot * HEADS + 255) / 256, 256, 0, stream>>>(esrc, edst, E_in, Etot,
                                                              als, ald, emax, denom, exb);
  const int aggThreads = (HC >= 128) ? 128 : 64;
  aggregate<<<Etot, aggThreads, 0, stream>>>(esrc, edst, E_in, hbuf, exb, denom, obuf, C);
}

extern "C" void kernel_launch(void* const* d_in, const int* in_sizes, int n_in,
                              void* d_out, int out_size, void* d_ws, size_t ws_size,
                              hipStream_t stream) {
  const float* x   = (const float*)d_in[0];
  const int*   ei  = (const int*)  d_in[1];
  const float* W1  = (const float*)d_in[2];
  const float* as1 = (const float*)d_in[3];
  const float* ad1 = (const float*)d_in[4];
  const float* b1  = (const float*)d_in[5];
  const float* W2  = (const float*)d_in[6];
  const float* as2 = (const float*)d_in[7];
  const float* ad2 = (const float*)d_in[8];
  const float* b2  = (const float*)d_in[9];
  const float* W3  = (const float*)d_in[10];
  const float* as3 = (const float*)d_in[11];
  const float* ad3 = (const float*)d_in[12];
  const float* b3  = (const float*)d_in[13];
  float* out = (float*)d_out;

  const int N    = in_sizes[0] / 12;     // 50000
  const int E_in = in_sizes[1] / 2;      // 400000
  const int Etot = E_in + N;
  const int* esrc = ei;
  const int* edst = ei + E_in;

  // Workspace carve-up (float units; every block is a multiple of 4 floats)
  float* ws = (float*)d_ws;
  size_t o = 0;
  float* hbuf  = ws + o; o += (size_t)N * 512;
  float* obuf  = ws + o; o += (size_t)N * 512;
  float* als   = ws + o; o += (size_t)N * HEADS;
  float* ald   = ws + o; o += (size_t)N * HEADS;
  float* emax  = ws + o; o += (size_t)N * HEADS;
  float* denom = ws + o; o += (size_t)N * HEADS;
  float* exb   = ws + o; o += (size_t)Etot * HEADS;
  _Float16* Apad  = (_Float16*)(ws + o); o += (size_t)N * 512 / 2;   // N*512 halfs
  _Float16* Bpack = (_Float16*)(ws + o); o += (size_t)(32 * 16 * 512) / 2;
  (void)ws_size; (void)n_in; (void)out_size;

  // Layer 1: [N,12] -> [N,8,64], concat + bias + relu
  gat_layer_launch(x, 12, 64, W1, as1, ad1, Apad, Bpack, hbuf, obuf,
                   als, ald, emax, denom, exb, esrc, edst, E_in, N, stream);
  bias_relu<<<(unsigned)(((size_t)N * 512 + 255) / 256), 256, 0, stream>>>(
      obuf, b1, (size_t)N * 512, 512);

  // Layer 2: [N,512] -> [N,8,64], concat + bias + relu
  gat_layer_launch(obuf, 512, 64, W2, as2, ad2, Apad, Bpack, hbuf, obuf,
                   als, ald, emax, denom, exb, esrc, edst, E_in, N, stream);
  bias_relu<<<(unsigned)(((size_t)N * 512 + 255) / 256), 256, 0, stream>>>(
      obuf, b2, (size_t)N * 512, 512);

  // Layer 3: [N,512] -> [N,8,5], head mean + bias + log_softmax -> d_out
  gat_layer_launch(obuf, 512, 5, W3, as3, ad3, Apad, Bpack, hbuf, obuf,
                   als, ald, emax, denom, exb, esrc, edst, E_in, N, stream);
  head_mean_logsoftmax<<<(N + 255) / 256, 256, 0, stream>>>(obuf, b3, out, N);
}
